// lstm_model_446676598985
// MI455X (gfx1250) — matrix operand order repaired and checked by reading, not compiled, not run on hardware
//
#include <hip/hip_runtime.h>

// ---------------------------------------------------------------------------
// CDNA5 (gfx1250, wave32) implementation of the bi-LSTM + LSTM + MHA model.
// All dense math uses v_wmma_f32_16x16x32_f16 (f16 A/B, f32 accumulate).
// ---------------------------------------------------------------------------

#define DEV __device__ __forceinline__

typedef __attribute__((ext_vector_type(8)))  _Float16 v8h;
typedef __attribute__((ext_vector_type(16))) _Float16 v16h;
typedef __attribute__((ext_vector_type(8)))  float    v8f;

DEV v16h cat8(v8h lo, v8h hi) {
  return __builtin_shufflevector(lo, hi, 0,1,2,3,4,5,6,7,8,9,10,11,12,13,14,15);
}

// A fragment (16x32 f16, row-major source, ld in elements).
// Layout: lanes 0-15 -> M=lane, K {0..15}; lanes 16-31 -> K {16..31}.
DEV v16h load_a_frag(const _Float16* base, int ld) {
  int l = threadIdx.x & 31;
  const _Float16* p = base + (l & 15) * ld + ((l >> 4) << 4);
  v8h lo = *(const v8h*)(p);
  v8h hi = *(const v8h*)(p + 8);
  return cat8(lo, hi);
}

// B fragment from pre-packed buffer: frag = 32 lanes x 16 f16 contiguous.
DEV v16h load_b_frag(const _Float16* packed) {
  int l = threadIdx.x & 31;
  const v8h* p = (const v8h*)(packed + (l << 4));
  return cat8(p[0], p[1]);
}

DEV v8f wmma16(v16h a, v16h b, v8f c) {
  return __builtin_amdgcn_wmma_f32_16x16x32_f16(false, a, false, b,
                                                (short)0, c, false, false);
}

DEV float sigm(float x)  { return 1.f / (1.f + __expf(-x)); }
DEV float tanh_(float x) { float e = __expf(2.f * x); return (e - 1.f) / (e + 1.f); }

DEV float block_sum(float v, float* red) {
  int t = threadIdx.x, n = blockDim.x;
  red[t] = v; __syncthreads();
  for (int s = n >> 1; s > 0; s >>= 1) {
    if (t < s) red[t] += red[t + s];
    __syncthreads();
  }
  float r = red[0]; __syncthreads();
  return r;
}

// ---------------------------------------------------------------------------
// Pack f32 weight [K,N] row-major into f16 WMMA B fragments.
// Fragment index = nt*KT + kt ; element (l,e) -> (k = kt*32+(l/16)*16+e, n = nt*16+l%16)
// ---------------------------------------------------------------------------
__global__ void k_pack_b(const float* __restrict__ W, _Float16* __restrict__ out,
                         int Korig, int N, int KT, int NT) {
  long idx = (long)blockIdx.x * 256 + threadIdx.x;
  long total = (long)KT * NT * 512;
  if (idx >= total) return;
  int  e    = (int)(idx & 15);
  int  l    = (int)((idx >> 4) & 31);
  long frag = idx >> 9;
  int  kt   = (int)(frag % KT);
  int  nt   = (int)(frag / KT);
  int  k    = kt * 32 + ((l >> 4) << 4) + e;
  int  n    = nt * 16 + (l & 15);
  float v = (k < Korig && n < N) ? W[(long)k * N + n] : 0.f;
  out[idx] = (_Float16)v;
}

// ---------------------------------------------------------------------------
// Input projection (F=8 -> 128) + LayerNorm, output f16. One block per (b,t).
// ---------------------------------------------------------------------------
__global__ void k_input_ln(const float* __restrict__ cgm, const float* __restrict__ Win,
                           const float* __restrict__ bin, const float* __restrict__ s,
                           const float* __restrict__ bb, _Float16* __restrict__ xf16) {
  __shared__ float red[128];
  int row = blockIdx.x, c = threadIdx.x;
  const float* xin = cgm + (long)row * 8;
  float a = bin[c];
#pragma unroll
  for (int k = 0; k < 8; k++) a += xin[k] * Win[k * 128 + c];
  float m   = block_sum(a, red) * (1.f / 128.f);
  float d   = a - m;
  float var = block_sum(d * d, red) * (1.f / 128.f);
  float y   = d * rsqrtf(var + 1e-5f) * s[c] + bb[c];
  xf16[(long)row * 128 + c] = (_Float16)y;
}

// ---------------------------------------------------------------------------
// Generic row LayerNorm: v = in*gate + preskip ; y = LN(v)*s + b + postskip
// One block per row, blockDim == N (128 or 256).
// ---------------------------------------------------------------------------
__global__ void k_ln(const float* __restrict__ in, const float* __restrict__ gate,
                     const float* __restrict__ preskip, const float* __restrict__ postskip,
                     const float* __restrict__ s, const float* __restrict__ b,
                     float* __restrict__ of32, _Float16* __restrict__ of16, int N) {
  __shared__ float red[256];
  long row = blockIdx.x;
  int  c   = threadIdx.x;
  long i   = row * N + c;
  float v = in[i];
  if (gate)    v *= gate[i];
  if (preskip) v += preskip[i];
  float invn = 1.f / (float)N;
  float m    = block_sum(v, red) * invn;
  float d    = v - m;
  float var  = block_sum(d * d, red) * invn;
  float y    = d * rsqrtf(var + 1e-5f) * s[c] + b[c];
  if (postskip) y += postskip[i];
  if (of32) of32[i] = y;
  if (of16) of16[i] = (_Float16)y;
}

// ---------------------------------------------------------------------------
// Generic WMMA GEMM: C[M,N] = act(A[M,K](f16) @ Bpacked + bias)
// 256 threads = 8 waves, one 16x16 tile per wave. act: 0=none 1=relu 2=sigmoid
// ---------------------------------------------------------------------------
__global__ void k_gemm(const _Float16* __restrict__ A, const _Float16* __restrict__ Bp,
                       const float* __restrict__ bias, float* __restrict__ Cf,
                       _Float16* __restrict__ Ch, int M, int N, int K, int act) {
  int KT = K >> 5, NT = N >> 4, MT = M >> 4;
  int tile = blockIdx.x * 8 + (threadIdx.x >> 5);
  if (tile >= MT * NT) return;             // wave-uniform
  int mt = tile / NT, nt = tile % NT;
  v8f acc;
#pragma unroll
  for (int r = 0; r < 8; r++) acc[r] = 0.f;
  const _Float16* Abase = A + (long)mt * 16 * K;
  const _Float16* Bbase = Bp + (long)nt * KT * 512;
  for (int kt = 0; kt < KT; ++kt) {
    if (kt + 1 < KT) __builtin_prefetch((const void*)(Bbase + (kt + 1) * 512), 0, 0);
    v16h a = load_a_frag(Abase + kt * 32, K);
    v16h b = load_b_frag(Bbase + kt * 512);
    acc = wmma16(a, b, acc);
  }
  int l  = threadIdx.x & 31;
  int n  = nt * 16 + (l & 15);
  float bv = bias ? bias[n] : 0.f;
#pragma unroll
  for (int r = 0; r < 8; r++) {
    int   m = mt * 16 + r + ((l >> 4) << 3);
    float v = acc[r] + bv;
    if (act == 1)      v = fmaxf(v, 0.f);
    else if (act == 2) v = sigm(v);
    long o = (long)m * N + n;
    if (Cf) Cf[o] = v;
    if (Ch) Ch[o] = (_Float16)v;
  }
}

// ---------------------------------------------------------------------------
// Bi-LSTM layer 0 (U=128). grid.x = 2 (dir). 1024 threads = 32 waves.
// z_t = h@Wh + x_t@Wi + b fused; h kept in LDS (f16), c in accumulator VGPRs.
// Wave w: mt = w&7 (batch tile), ng = w>>3; owns 16x32 patch per gate.
// ---------------------------------------------------------------------------
__global__ void __launch_bounds__(1024) k_lstm0(
    const _Float16* __restrict__ x,
    const _Float16* __restrict__ pWhF, const _Float16* __restrict__ pWiF,
    const float* __restrict__ bF,
    const _Float16* __restrict__ pWhB, const _Float16* __restrict__ pWiB,
    const float* __restrict__ bB,
    float* __restrict__ catOut) {
  __shared__ _Float16 hsh[128 * 128];
  int dir = blockIdx.x;
  const _Float16* pWh = dir ? pWhB : pWhF;
  const _Float16* pWi = dir ? pWiB : pWiF;
  const float*   bias = dir ? bB   : bF;
  int tid = threadIdx.x, l = tid & 31, wid = tid >> 5;
  int mt = wid & 7, ng = wid >> 3;
  for (int i = tid; i < 128 * 128; i += 1024) hsh[i] = (_Float16)0.f;
  float c[2][8];
#pragma unroll
  for (int j = 0; j < 2; j++)
#pragma unroll
    for (int r = 0; r < 8; r++) c[j][r] = 0.f;
  __syncthreads();

  for (int t = 0; t < 288; t++) {
    int tt = dir ? (287 - t) : t;
    v8f acc[4][2];
#pragma unroll
    for (int q = 0; q < 4; q++)
#pragma unroll
      for (int j = 0; j < 2; j++)
#pragma unroll
        for (int r = 0; r < 8; r++) acc[q][j][r] = 0.f;

    for (int kt = 0; kt < 4; kt++) {                 // h @ Wh  (K=128)
      v16h a = load_a_frag(hsh + mt * 16 * 128 + kt * 32, 128);
#pragma unroll
      for (int q = 0; q < 4; q++)
#pragma unroll
        for (int j = 0; j < 2; j++) {
          int nt = q * 8 + ng * 2 + j;
          acc[q][j] = wmma16(a, load_b_frag(pWh + (long)(nt * 4 + kt) * 512), acc[q][j]);
        }
    }
    for (int kt = 0; kt < 4; kt++) {                 // x_t @ Wi (K=128)
      v16h a = load_a_frag(x + (long)(mt * 16) * (288 * 128) + (long)tt * 128 + kt * 32,
                           288 * 128);
#pragma unroll
      for (int q = 0; q < 4; q++)
#pragma unroll
        for (int j = 0; j < 2; j++) {
          int nt = q * 8 + ng * 2 + j;
          acc[q][j] = wmma16(a, load_b_frag(pWi + (long)(nt * 4 + kt) * 512), acc[q][j]);
        }
    }

    float hv[2][8];
#pragma unroll
    for (int j = 0; j < 2; j++) {
      int ncol = ng * 32 + j * 16 + (l & 15);
      float bi = bias[ncol], bfv = bias[128 + ncol];
      float bg = bias[256 + ncol], bo = bias[384 + ncol];
#pragma unroll
      for (int r = 0; r < 8; r++) {
        float iv = sigm(acc[0][j][r] + bi);
        float fv = sigm(acc[1][j][r] + bfv);
        float gv = tanh_(acc[2][j][r] + bg);
        float ov = sigm(acc[3][j][r] + bo);
        float cc = fv * c[j][r] + iv * gv;
        c[j][r] = cc;
        float h = ov * tanh_(cc);
        hv[j][r] = h;
        int m = mt * 16 + r + ((l >> 4) << 3);
        catOut[((long)m * 288 + tt) * 256 + dir * 128 + ncol] = h;
      }
    }
    __syncthreads();                                  // old h fully consumed
#pragma unroll
    for (int j = 0; j < 2; j++) {
      int ncol = ng * 32 + j * 16 + (l & 15);
#pragma unroll
      for (int r = 0; r < 8; r++) {
        int m = mt * 16 + r + ((l >> 4) << 3);
        hsh[m * 128 + ncol] = (_Float16)hv[j][r];
      }
    }
    __syncthreads();                                  // new h visible
  }
}

// ---------------------------------------------------------------------------
// LSTM layer 1 (U=256). 1 block, 1024 threads = 32 waves.
// h in LDS (128x256 f16 = 64KB); Wh/Wi streamed as packed fragments from L2.
// Wave w: mt = w&7, ng = w>>3; owns 16x64 patch per gate (acc[4][4]).
// ---------------------------------------------------------------------------
__global__ void __launch_bounds__(1024) k_lstm1(
    const _Float16* __restrict__ x, const _Float16* __restrict__ pWh,
    const _Float16* __restrict__ pWi, const float* __restrict__ bias,
    float* __restrict__ outp) {
  __shared__ _Float16 hsh[128 * 256];
  int tid = threadIdx.x, l = tid & 31, wid = tid >> 5;
  int mt = wid & 7, ng = wid >> 3;
  for (int i = tid; i < 128 * 256; i += 1024) hsh[i] = (_Float16)0.f;
  float c[4][8];
#pragma unroll
  for (int j = 0; j < 4; j++)
#pragma unroll
    for (int r = 0; r < 8; r++) c[j][r] = 0.f;
  __syncthreads();

  for (int t = 0; t < 288; t++) {
    v8f acc[4][4];
#pragma unroll
    for (int q = 0; q < 4; q++)
#pragma unroll
      for (int j = 0; j < 4; j++)
#pragma unroll
        for (int r = 0; r < 8; r++) acc[q][j][r] = 0.f;

    for (int kt = 0; kt < 8; kt++) {                 // h @ Wh  (K=256)
      v16h a = load_a_frag(hsh + mt * 16 * 256 + kt * 32, 256);
#pragma unroll
      for (int q = 0; q < 4; q++)
#pragma unroll
        for (int j = 0; j < 4; j++) {
          int nt = q * 16 + ng * 4 + j;
          acc[q][j] = wmma16(a, load_b_frag(pWh + (long)(nt * 8 + kt) * 512), acc[q][j]);
        }
    }
    for (int kt = 0; kt < 8; kt++) {                 // x_t @ Wi (K=256)
      v16h a = load_a_frag(x + (long)(mt * 16) * (288 * 256) + (long)t * 256 + kt * 32,
                           288 * 256);
#pragma unroll
      for (int q = 0; q < 4; q++)
#pragma unroll
        for (int j = 0; j < 4; j++) {
          int nt = q * 16 + ng * 4 + j;
          acc[q][j] = wmma16(a, load_b_frag(pWi + (long)(nt * 8 + kt) * 512), acc[q][j]);
        }
    }

    float hv[4][8];
#pragma unroll
    for (int j = 0; j < 4; j++) {
      int ncol = ng * 64 + j * 16 + (l & 15);
      float bi = bias[ncol], bfv = bias[256 + ncol];
      float bg = bias[512 + ncol], bo = bias[768 + ncol];
#pragma unroll
      for (int r = 0; r < 8; r++) {
        float iv = sigm(acc[0][j][r] + bi);
        float fv = sigm(acc[1][j][r] + bfv);
        float gv = tanh_(acc[2][j][r] + bg);
        float ov = sigm(acc[3][j][r] + bo);
        float cc = fv * c[j][r] + iv * gv;
        c[j][r] = cc;
        float h = ov * tanh_(cc);
        hv[j][r] = h;
        int m = mt * 16 + r + ((l >> 4) << 3);
        outp[((long)m * 288 + t) * 256 + ncol] = h;
      }
    }
    __syncthreads();
#pragma unroll
    for (int j = 0; j < 4; j++) {
      int ncol = ng * 64 + j * 16 + (l & 15);
#pragma unroll
      for (int r = 0; r < 8; r++) {
        int m = mt * 16 + r + ((l >> 4) << 3);
        hsh[m * 256 + ncol] = (_Float16)hv[j][r];
      }
    }
    __syncthreads();
  }
}

// ---------------------------------------------------------------------------
// Attention: one block per (b, head, q-tile). 256 threads = 8 waves.
// scores (WMMA, K=64) -> softmax (LDS) -> attn@V (WMMA, K=288) -> mhacat f16.
// ---------------------------------------------------------------------------
__global__ void k_attn(const _Float16* __restrict__ q, const _Float16* __restrict__ k,
                       const _Float16* __restrict__ v, _Float16* __restrict__ mhacat) {
  __shared__ float    sc[16 * 288];
  __shared__ _Float16 at[16 * 288];
  __shared__ float    red2[256];
  int qt = blockIdx.x % 18;
  int hd = (blockIdx.x / 18) & 3;
  int b  = blockIdx.x / 72;
  int tid = threadIdx.x, l = tid & 31, wid = tid >> 5;
  const _Float16* qb = q + ((long)b * 288 + qt * 16) * 256 + hd * 64;

  // phase 1: scores = q . k^T * 1/sqrt(64)
  for (int nt = wid; nt < 18; nt += 8) {
    v8f acc;
#pragma unroll
    for (int r = 0; r < 8; r++) acc[r] = 0.f;
#pragma unroll
    for (int kt = 0; kt < 2; kt++) {
      v16h a = load_a_frag(qb + kt * 32, 256);
      const _Float16* kp = k + ((long)b * 288 + nt * 16 + (l & 15)) * 256 + hd * 64 +
                           kt * 32 + ((l >> 4) << 4);
      v8h lo = *(const v8h*)(kp);
      v8h hi = *(const v8h*)(kp + 8);
      acc = wmma16(a, cat8(lo, hi), acc);
    }
#pragma unroll
    for (int r = 0; r < 8; r++) {
      int m = r + ((l >> 4) << 3);
      sc[m * 288 + nt * 16 + (l & 15)] = acc[r] * 0.125f;
    }
  }
  __syncthreads();

  // phase 2: softmax rows (16 threads / row)
  int rr = tid >> 4, rl = tid & 15;
  float mx = -1e30f;
  for (int ci = rl; ci < 288; ci += 16) mx = fmaxf(mx, sc[rr * 288 + ci]);
  red2[tid] = mx; __syncthreads();
#pragma unroll
  for (int st = 8; st > 0; st >>= 1) {
    if (rl < st) red2[tid] = fmaxf(red2[tid], red2[tid + st]);
    __syncthreads();
  }
  mx = red2[rr * 16]; __syncthreads();
  float sum = 0.f;
  for (int ci = rl; ci < 288; ci += 16) {
    float e = __expf(sc[rr * 288 + ci] - mx);
    sc[rr * 288 + ci] = e;
    sum += e;
  }
  red2[tid] = sum; __syncthreads();
#pragma unroll
  for (int st = 8; st > 0; st >>= 1) {
    if (rl < st) red2[tid] += red2[tid + st];
    __syncthreads();
  }
  float inv = 1.f / red2[rr * 16]; __syncthreads();
  for (int ci = rl; ci < 288; ci += 16)
    at[rr * 288 + ci] = (_Float16)(sc[rr * 288 + ci] * inv);
  __syncthreads();

  // phase 3: out = attn @ V  (waves 0..3, nt2 = dh tile)
  if (wid < 4) {
    int nt2 = wid;
    v8f acc;
#pragma unroll
    for (int r = 0; r < 8; r++) acc[r] = 0.f;
    for (int kt = 0; kt < 9; kt++) {
      v16h a = load_a_frag(at + kt * 32, 288);
      const _Float16* vp = v + ((long)b * 288 + kt * 32 + ((l >> 4) << 4)) * 256 +
                           hd * 64 + nt2 * 16 + (l & 15);
      v16h bf;
#pragma unroll
      for (int e = 0; e < 16; e++) bf[e] = vp[(long)e * 256];
      acc = wmma16(a, bf, acc);
    }
#pragma unroll
    for (int r = 0; r < 8; r++) {
      int m = r + ((l >> 4) << 3);
      mhacat[((long)b * 288 + qt * 16 + m) * 256 + hd * 64 + nt2 * 16 + (l & 15)] =
          (_Float16)acc[r];
    }
  }
}

// ---------------------------------------------------------------------------
// Pool (mean+max over T) + static features -> z f16 [128, 544] (528 + pad).
// ---------------------------------------------------------------------------
__global__ void k_pool(const float* __restrict__ y2, const float* __restrict__ other,
                       _Float16* __restrict__ z) {
  int b = blockIdx.x, c = threadIdx.x;
  const float* p = y2 + (long)b * 288 * 256 + c;
  float s = 0.f, mx = -1e30f;
  for (int t = 0; t < 288; t++) {
    float vv = p[(long)t * 256];
    s += vv;
    mx = fmaxf(mx, vv);
  }
  z[(long)b * 544 + c]       = (_Float16)(s * (1.f / 288.f));
  z[(long)b * 544 + 256 + c] = (_Float16)mx;
  if (c < 32) {
    float vv = (c < 16) ? other[b * 16 + c] : 0.f;
    z[(long)b * 544 + 512 + c] = (_Float16)vv;
  }
}

// ---------------------------------------------------------------------------
// Final projection: out[b] = h2n[b,:] . W_out + b_out
// ---------------------------------------------------------------------------
__global__ void k_final(const float* __restrict__ h2, const float* __restrict__ W,
                        const float* __restrict__ bo, float* __restrict__ out) {
  int b = threadIdx.x;
  if (b >= 128) return;
  float s = bo[0];
  for (int i = 0; i < 128; i++) s += h2[b * 128 + i] * W[i];
  out[b] = s;
}

// ---------------------------------------------------------------------------
extern "C" void kernel_launch(void* const* d_in, const int* in_sizes, int n_in,
                              void* d_out, int out_size, void* d_ws, size_t ws_size,
                              hipStream_t stream) {
  (void)in_sizes; (void)n_in; (void)out_size; (void)ws_size;
  const float* cgm     = (const float*)d_in[0];
  const float* other   = (const float*)d_in[1];
  const float* W_in    = (const float*)d_in[2];
  const float* b_in    = (const float*)d_in[3];
  const float* ln_in_s = (const float*)d_in[4];
  const float* ln_in_b = (const float*)d_in[5];
  const float* Wi_f    = (const float*)d_in[6];
  const float* Wh_f    = (const float*)d_in[7];
  const float* b_f     = (const float*)d_in[8];
  const float* Wi_b    = (const float*)d_in[9];
  const float* Wh_b    = (const float*)d_in[10];
  const float* b_b     = (const float*)d_in[11];
  const float* ln_bi_s = (const float*)d_in[12];
  const float* ln_bi_b = (const float*)d_in[13];
  const float* Wi_a    = (const float*)d_in[14];
  const float* Wh_a    = (const float*)d_in[15];
  const float* b_a     = (const float*)d_in[16];
  const float* ln_a_s  = (const float*)d_in[17];
  const float* ln_a_b  = (const float*)d_in[18];
  const float* Wq = (const float*)d_in[19]; const float* bq = (const float*)d_in[20];
  const float* Wk = (const float*)d_in[21]; const float* bk = (const float*)d_in[22];
  const float* Wv = (const float*)d_in[23]; const float* bv = (const float*)d_in[24];
  const float* Wo = (const float*)d_in[25]; const float* bo = (const float*)d_in[26];
  const float* Wg = (const float*)d_in[27]; const float* bg = (const float*)d_in[28];
  const float* ln_blk_s = (const float*)d_in[29];
  const float* ln_blk_b = (const float*)d_in[30];
  const float* W1 = (const float*)d_in[31]; const float* b1 = (const float*)d_in[32];
  const float* ln_d1_s = (const float*)d_in[33];
  const float* ln_d1_b = (const float*)d_in[34];
  const float* W2 = (const float*)d_in[35]; const float* b2 = (const float*)d_in[36];
  const float* ln_d2_s = (const float*)d_in[37];
  const float* ln_d2_b = (const float*)d_in[38];
  const float* W_out = (const float*)d_in[39];
  const float* b_out = (const float*)d_in[40];
  float* out = (float*)d_out;

  char* wsb = (char*)d_ws;
  size_t off = 0;
  auto alloc = [&](size_t nbytes) -> char* {
    char* p = wsb + off;
    off += (nbytes + 255) & ~(size_t)255;
    return p;
  };
  auto packw = [&](const float* W, int K, int N) -> _Float16* {
    int KT = (K + 31) / 32, NT = N / 16;
    _Float16* p = (_Float16*)alloc((size_t)KT * NT * 1024);
    long total = (long)KT * NT * 512;
    k_pack_b<<<dim3((unsigned)((total + 255) / 256)), dim3(256), 0, stream>>>(W, p, K, N, KT, NT);
    return p;
  };

  // --- pack all weights into WMMA B-fragment layout (f16) ---
  _Float16* pWiF = packw(Wi_f, 128, 512);
  _Float16* pWhF = packw(Wh_f, 128, 512);
  _Float16* pWiB = packw(Wi_b, 128, 512);
  _Float16* pWhB = packw(Wh_b, 128, 512);
  _Float16* pWiA = packw(Wi_a, 256, 1024);
  _Float16* pWhA = packw(Wh_a, 256, 1024);
  _Float16* pWq  = packw(Wq, 256, 256);
  _Float16* pWk  = packw(Wk, 256, 256);
  _Float16* pWv  = packw(Wv, 256, 256);
  _Float16* pWo  = packw(Wo, 256, 256);
  _Float16* pWg  = packw(Wg, 256, 256);
  _Float16* pW1  = packw(W1, 528, 256);   // zero-padded to K=544
  _Float16* pW2  = packw(W2, 256, 128);

  // --- activation scratch (with lifetime-based reuse) ---
  const long M = 128L * 288;   // 36864 rows
  _Float16* xf16  = (_Float16*)alloc(M * 128 * 2);
  float*    buf0  = (float*)alloc(M * 256 * 4);   // concat -> attn_out
  float*    buf1  = (float*)alloc(M * 256 * 4);   // x1     -> y2
  float*    buf2  = (float*)alloc(M * 256 * 4);   // lstm1  -> gate
  float*    bufY  = (float*)alloc(M * 256 * 4);   // y
  _Float16* x1f16 = (_Float16*)alloc(M * 256 * 2);  // x1 f16 -> mhacat
  _Float16* yf16  = (_Float16*)alloc(M * 256 * 2);
  _Float16* qf16  = (_Float16*)alloc(M * 256 * 2);
  _Float16* kf16  = (_Float16*)alloc(M * 256 * 2);
  _Float16* vf16  = (_Float16*)alloc(M * 256 * 2);
  _Float16* zf16  = (_Float16*)alloc(128L * 544 * 2);
  float*    h1    = (float*)alloc(128L * 256 * 4);
  _Float16* h1f16 = (_Float16*)alloc(128L * 256 * 2);
  float*    h2    = (float*)alloc(128L * 128 * 4);
  float*    h2n   = (float*)alloc(128L * 128 * 4);

  auto gemm = [&](const _Float16* A, const _Float16* Bp, const float* bias,
                  float* Cf, _Float16* Ch, int Mm, int N, int K, int act) {
    int tiles = (Mm / 16) * (N / 16);
    k_gemm<<<dim3((unsigned)((tiles + 7) / 8)), dim3(256), 0, stream>>>(
        A, Bp, bias, Cf, Ch, Mm, N, K, act);
  };
  auto ln = [&](const float* in, const float* gate, const float* pre, const float* post,
                const float* s, const float* b, float* of, _Float16* oh, int Mm, int N) {
    k_ln<<<dim3((unsigned)Mm), dim3((unsigned)N), 0, stream>>>(in, gate, pre, post, s, b,
                                                               of, oh, N);
  };

  // 1. input projection + LN -> x (f16)
  k_input_ln<<<dim3(36864), dim3(128), 0, stream>>>(cgm, W_in, b_in, ln_in_s, ln_in_b, xf16);
  // 2. bidirectional LSTM layer 0 -> concat [B,T,256] (f32) in buf0
  k_lstm0<<<dim3(2), dim3(1024), 0, stream>>>(xf16, pWhF, pWiF, b_f, pWhB, pWiB, b_b, buf0);
  // 3. concat LayerNorm -> x1 (f32 buf1, f16 x1f16)
  ln(buf0, nullptr, nullptr, nullptr, ln_bi_s, ln_bi_b, buf1, x1f16, 36864, 256);
  // 4. LSTM layer 1 -> buf2
  k_lstm1<<<dim3(1), dim3(1024), 0, stream>>>(x1f16, pWhA, pWiA, b_a, buf2);
  // 5. y = LN(lstm1) + x1 -> bufY (+ yf16)
  ln(buf2, nullptr, nullptr, buf1, ln_a_s, ln_a_b, bufY, yf16, 36864, 256);
  // 6. q,k,v projections (f16) and gate = sigmoid(y@Wg+bg) (f32 in buf2)
  gemm(yf16, pWq, bq, nullptr, qf16, 36864, 256, 256, 0);
  gemm(yf16, pWk, bk, nullptr, kf16, 36864, 256, 256, 0);
  gemm(yf16, pWv, bv, nullptr, vf16, 36864, 256, 256, 0);
  gemm(yf16, pWg, bg, buf2, nullptr, 36864, 256, 256, 2);
  // 7. multi-head attention -> mhacat (reuse x1f16)
  k_attn<<<dim3(128 * 4 * 18), dim3(256), 0, stream>>>(qf16, kf16, vf16, x1f16);
  // 8. output projection -> attn_out (reuse buf0)
  gemm(x1f16, pWo, bo, buf0, nullptr, 36864, 256, 256, 0);
  // 9. y2 = LN(attn_out * gate + y) -> buf1
  ln(buf0, buf2, bufY, nullptr, ln_blk_s, ln_blk_b, buf1, nullptr, 36864, 256);
  // 10. pooling + static features -> z (f16, K padded to 544)
  k_pool<<<dim3(128), dim3(256), 0, stream>>>(buf1, other, zf16);
  // 11. MLP head
  gemm(zf16, pW1, b1, h1, nullptr, 128, 256, 544, 1);
  ln(h1, nullptr, nullptr, nullptr, ln_d1_s, ln_d1_b, nullptr, h1f16, 128, 256);
  gemm(h1f16, pW2, b2, h2, nullptr, 128, 128, 256, 1);
  ln(h2, nullptr, nullptr, nullptr, ln_d2_s, ln_d2_b, h2n, nullptr, 128, 128);
  k_final<<<dim3(1), dim3(128), 0, stream>>>(h2n, W_out, b_out, out);
}
